// CausalSelfAttention_77781857730836
// MI455X (gfx1250) — matrix operand order repaired and checked by reading, not compiled, not run on hardware
//
#include <hip/hip_runtime.h>
#include <hip/hip_bf16.h>

typedef __attribute__((ext_vector_type(16))) __bf16 v16bf;
typedef __attribute__((ext_vector_type(8)))  __bf16 v8bf;
typedef __attribute__((ext_vector_type(8)))  float  v8f;

#define D_MODEL 1024
#define N_HEADS 16
#define D_HEADD 64
#define BATCH   4
#define SEQ     2048
#define M_TOTAL (BATCH * SEQ)      /* 8192 */
#define N_QKV   (3 * D_MODEL)      /* 3072 */

// ---------- WMMA helper: D = A(16x32 bf16) * B(32x16 bf16) + C(16x16 f32) ----------
__device__ __forceinline__ v8f wmma_bf16(v16bf a, v16bf b, v8f c) {
  return __builtin_amdgcn_wmma_f32_16x16x32_bf16(false, a, false, b, (short)0, c, false, false);
}

// A-fragment, 16x32, from row-major [rows, ld]: lane holds row (lane&15),
// K chunks [8*half, 8*half+8) and [16+8*half, 16+8*half+8)  (ISA 7.12.2)
__device__ __forceinline__ v16bf load_a_frag(const __bf16* base, int ld, int row0, int k0, int lane) {
  int r  = row0 + (lane & 15);
  int kk = k0 + ((lane >> 4) << 3);
  const __bf16* p = base + (size_t)r * ld + kk;
  union { v16bf v; v8bf h[2]; } u;
  u.h[0] = *(const v8bf*)(p);
  u.h[1] = *(const v8bf*)(p + 16);
  return u.v;
}

// B-fragment, 32x16, from N-major storage [n, ld(=K)]: lane holds col (lane&15),
// K chunks [8*half, 8*half+8) and [16+8*half, 16+8*half+8) (same order as A)
__device__ __forceinline__ v16bf load_b_frag(const __bf16* base, int ld, int n0, int k0, int lane) {
  int n  = n0 + (lane & 15);
  int kk = k0 + ((lane >> 4) << 3);
  const __bf16* p = base + (size_t)n * ld + kk;
  union { v16bf v; v8bf h[2]; } u;
  u.h[0] = *(const v8bf*)(p);
  u.h[1] = *(const v8bf*)(p + 16);
  return u.v;
}

// ---------------- pack kernels ----------------
__global__ void cvt_bf16_kernel(const float* __restrict__ in, __bf16* __restrict__ out, int n) {
  int i = blockIdx.x * blockDim.x + threadIdx.x;
  if (i < n) out[i] = (__bf16)in[i];
}

// in [rows, cols] f32 row-major -> out [cols, rows] bf16 (N-major weights)
__global__ void transpose_cvt_kernel(const float* __restrict__ in, __bf16* __restrict__ out,
                                     int rows, int cols) {
  int i = blockIdx.x * blockDim.x + threadIdx.x;
  if (i < rows * cols) {
    int r = i / cols, c = i % cols;
    out[(size_t)c * rows + r] = (__bf16)in[i];
  }
}

// ---------------- QKV projection: [8192,1024] @ [1024,3072] ----------------
// Wave computes 32x64 tile. Scatters to Q,K [B,H,T,Dh] and V^T [B,H,Dh,T] (all bf16).
__global__ __launch_bounds__(256) void qkv_gemm_kernel(
    const __bf16* __restrict__ xb, const __bf16* __restrict__ wT,
    const float* __restrict__ bias,
    __bf16* __restrict__ q_ws, __bf16* __restrict__ k_ws, __bf16* __restrict__ vt_ws) {
  int wave = (blockIdx.x * blockDim.x + threadIdx.x) >> 5;
  int lane = threadIdx.x & 31;
  const int nTiles = N_QKV / 64;                 // 48
  int m0 = (wave / nTiles) * 32;
  int n0 = (wave % nTiles) * 64;

  v8f z = {0.f,0.f,0.f,0.f,0.f,0.f,0.f,0.f};
  v8f acc[2][4];
  for (int i = 0; i < 2; ++i) for (int j = 0; j < 4; ++j) acc[i][j] = z;

  for (int k0 = 0; k0 < D_MODEL; k0 += 32) {
    v16bf a0 = load_a_frag(xb, D_MODEL, m0,      k0, lane);
    v16bf a1 = load_a_frag(xb, D_MODEL, m0 + 16, k0, lane);
    v16bf b0 = load_b_frag(wT, D_MODEL, n0,      k0, lane);
    v16bf b1 = load_b_frag(wT, D_MODEL, n0 + 16, k0, lane);
    v16bf b2 = load_b_frag(wT, D_MODEL, n0 + 32, k0, lane);
    v16bf b3 = load_b_frag(wT, D_MODEL, n0 + 48, k0, lane);
    acc[0][0] = wmma_bf16(a0, b0, acc[0][0]);
    acc[0][1] = wmma_bf16(a0, b1, acc[0][1]);
    acc[0][2] = wmma_bf16(a0, b2, acc[0][2]);
    acc[0][3] = wmma_bf16(a0, b3, acc[0][3]);
    acc[1][0] = wmma_bf16(a1, b0, acc[1][0]);
    acc[1][1] = wmma_bf16(a1, b1, acc[1][1]);
    acc[1][2] = wmma_bf16(a1, b2, acc[1][2]);
    acc[1][3] = wmma_bf16(a1, b3, acc[1][3]);
  }

  int col = lane & 15, hf = lane >> 4;
  for (int i = 0; i < 2; ++i) {
    for (int j = 0; j < 4; ++j) {
      int n = n0 + j * 16 + col;
      float bv = bias[n];
      int which = n >> 10;           // 0:Q 1:K 2:V
      int nn = n & 1023;
      int h = nn >> 6, d = nn & 63;
      for (int r = 0; r < 8; ++r) {
        int m = m0 + i * 16 + r + hf * 8;
        int b = m >> 11, t = m & 2047;
        __bf16 o = (__bf16)(acc[i][j][r] + bv);
        int bh = b * N_HEADS + h;
        if (which == 0)      q_ws [((size_t)bh * SEQ + t) * D_HEADD + d] = o;
        else if (which == 1) k_ws [((size_t)bh * SEQ + t) * D_HEADD + d] = o;
        else                 vt_ws[((size_t)bh * D_HEADD + d) * SEQ + t] = o;
      }
    }
  }
}

// ---------------- flash attention: one wave = 16 queries of one (b,h) ----------------
__global__ __launch_bounds__(128) void attn_kernel(
    const __bf16* __restrict__ q_ws, const __bf16* __restrict__ k_ws,
    const __bf16* __restrict__ vt_ws, __bf16* __restrict__ y_ws) {
  __shared__ __bf16 pbuf[4][16][40];             // P tile per wave, 80B rows (16B aligned)
  int w = threadIdx.x >> 5, lane = threadIdx.x & 31;
  int bh = blockIdx.y;
  int q0 = (blockIdx.x * 4 + w) * 16;
  const __bf16* qb = q_ws  + (size_t)bh * SEQ * D_HEADD;
  const __bf16* kb = k_ws  + (size_t)bh * SEQ * D_HEADD;
  const __bf16* vb = vt_ws + (size_t)bh * D_HEADD * SEQ;

  v16bf aq0 = load_a_frag(qb, D_HEADD, q0, 0,  lane);
  v16bf aq1 = load_a_frag(qb, D_HEADD, q0, 32, lane);

  v8f z = {0.f,0.f,0.f,0.f,0.f,0.f,0.f,0.f};
  v8f acc[4] = {z, z, z, z};
  float mrow[8], lrow[8];
  for (int r = 0; r < 8; ++r) { mrow[r] = -1e30f; lrow[r] = 0.f; }

  int col = lane & 15, hf = lane >> 4;
  int nkb = (q0 + 15) / 32 + 1;                  // causal key-block count
  for (int kbk = 0; kbk < nkb; ++kbk) {
    int j0 = kbk * 32;
    v8f s0 = z, s1 = z;
    s0 = wmma_bf16(aq0, load_b_frag(kb, D_HEADD, j0,      0,  lane), s0);
    s0 = wmma_bf16(aq1, load_b_frag(kb, D_HEADD, j0,      32, lane), s0);
    s1 = wmma_bf16(aq0, load_b_frag(kb, D_HEADD, j0 + 16, 0,  lane), s1);
    s1 = wmma_bf16(aq1, load_b_frag(kb, D_HEADD, j0 + 16, 32, lane), s1);

    float p0[8], p1[8], alpha[8];
    for (int r = 0; r < 8; ++r) {
      int qrow = q0 + r + hf * 8;
      float x0 = s0[r] * 0.125f;                 // 1/sqrt(64)
      float x1 = s1[r] * 0.125f;
      if (j0 + col      > qrow) x0 = -1e30f;     // causal mask
      if (j0 + 16 + col > qrow) x1 = -1e30f;
      float tmax = fmaxf(x0, x1);
      for (int off = 1; off < 16; off <<= 1)
        tmax = fmaxf(tmax, __shfl_xor(tmax, off, 32));
      float mn = fmaxf(mrow[r], tmax);
      float a  = __expf(mrow[r] - mn);
      float e0 = __expf(x0 - mn), e1 = __expf(x1 - mn);
      float rs = e0 + e1;
      for (int off = 1; off < 16; off <<= 1)
        rs += __shfl_xor(rs, off, 32);
      lrow[r] = lrow[r] * a + rs;
      mrow[r] = mn;
      alpha[r] = a; p0[r] = e0; p1[r] = e1;
    }
    for (int j = 0; j < 4; ++j)
      for (int r = 0; r < 8; ++r) acc[j][r] *= alpha[r];

    // C-layout -> A-layout for P via this wave's private LDS tile
    for (int r = 0; r < 8; ++r) {
      int row = r + hf * 8;
      pbuf[w][row][col]      = (__bf16)p0[r];
      pbuf[w][row][16 + col] = (__bf16)p1[r];
    }
    asm volatile("s_wait_dscnt 0" ::: "memory"); // DS in-order per wave; all stores visible
    v16bf ap = load_a_frag(&pbuf[w][0][0], 40, 0, 0, lane);

    for (int j = 0; j < 4; ++j)
      acc[j] = wmma_bf16(ap, load_b_frag(vb, SEQ, j * 16, j0, lane), acc[j]);
  }

  int b = bh / N_HEADS, h = bh % N_HEADS;
  for (int j = 0; j < 4; ++j)
    for (int r = 0; r < 8; ++r) {
      int t = q0 + r + hf * 8;
      int c = h * D_HEADD + j * 16 + col;
      y_ws[((size_t)(b * SEQ + t)) * D_MODEL + c] = (__bf16)(acc[j][r] / lrow[r]);
    }
}

// ---------------- output projection: [8192,1024] @ [1024,1024] + bias -> f32 ----------------
__global__ __launch_bounds__(256) void out_gemm_kernel(
    const __bf16* __restrict__ yb, const __bf16* __restrict__ wT,
    const float* __restrict__ bias, float* __restrict__ out) {
  int wave = (blockIdx.x * blockDim.x + threadIdx.x) >> 5;
  int lane = threadIdx.x & 31;
  const int nTiles = D_MODEL / 64;               // 16
  int m0 = (wave / nTiles) * 32;
  int n0 = (wave % nTiles) * 64;

  v8f z = {0.f,0.f,0.f,0.f,0.f,0.f,0.f,0.f};
  v8f acc[2][4];
  for (int i = 0; i < 2; ++i) for (int j = 0; j < 4; ++j) acc[i][j] = z;

  for (int k0 = 0; k0 < D_MODEL; k0 += 32) {
    v16bf a0 = load_a_frag(yb, D_MODEL, m0,      k0, lane);
    v16bf a1 = load_a_frag(yb, D_MODEL, m0 + 16, k0, lane);
    v16bf b0 = load_b_frag(wT, D_MODEL, n0,      k0, lane);
    v16bf b1 = load_b_frag(wT, D_MODEL, n0 + 16, k0, lane);
    v16bf b2 = load_b_frag(wT, D_MODEL, n0 + 32, k0, lane);
    v16bf b3 = load_b_frag(wT, D_MODEL, n0 + 48, k0, lane);
    acc[0][0] = wmma_bf16(a0, b0, acc[0][0]);
    acc[0][1] = wmma_bf16(a0, b1, acc[0][1]);
    acc[0][2] = wmma_bf16(a0, b2, acc[0][2]);
    acc[0][3] = wmma_bf16(a0, b3, acc[0][3]);
    acc[1][0] = wmma_bf16(a1, b0, acc[1][0]);
    acc[1][1] = wmma_bf16(a1, b1, acc[1][1]);
    acc[1][2] = wmma_bf16(a1, b2, acc[1][2]);
    acc[1][3] = wmma_bf16(a1, b3, acc[1][3]);
  }

  int col = lane & 15, hf = lane >> 4;
  for (int i = 0; i < 2; ++i)
    for (int j = 0; j < 4; ++j) {
      int n = n0 + j * 16 + col;
      float bv = bias[n];
      for (int r = 0; r < 8; ++r) {
        int m = m0 + i * 16 + r + hf * 8;
        out[(size_t)m * D_MODEL + n] = acc[i][j][r] + bv;
      }
    }
}

// ---------------- host ----------------
extern "C" void kernel_launch(void* const* d_in, const int* in_sizes, int n_in,
                              void* d_out, int out_size, void* d_ws, size_t ws_size,
                              hipStream_t stream) {
  const float* x     = (const float*)d_in[0];
  const float* W_qkv = (const float*)d_in[1];
  const float* b_qkv = (const float*)d_in[2];
  const float* W_out = (const float*)d_in[3];
  const float* b_out = (const float*)d_in[4];
  float* out = (float*)d_out;

  char* ws = (char*)d_ws;
  size_t off = 0;
  __bf16* xb     = (__bf16*)(ws + off); off += (size_t)M_TOTAL * D_MODEL * 2;   // 16 MB
  __bf16* wqkvT  = (__bf16*)(ws + off); off += (size_t)N_QKV * D_MODEL * 2;     //  6 MB
  __bf16* woutT  = (__bf16*)(ws + off); off += (size_t)D_MODEL * D_MODEL * 2;   //  2 MB
  __bf16* q_ws   = (__bf16*)(ws + off); off += (size_t)M_TOTAL * D_MODEL * 2;   // 16 MB
  __bf16* k_ws   = (__bf16*)(ws + off); off += (size_t)M_TOTAL * D_MODEL * 2;   // 16 MB
  __bf16* vt_ws  = (__bf16*)(ws + off); off += (size_t)M_TOTAL * D_MODEL * 2;   // 16 MB
  __bf16* y_ws   = (__bf16*)(ws + off); off += (size_t)M_TOTAL * D_MODEL * 2;   // 16 MB
  (void)ws_size; (void)n_in; (void)in_sizes; (void)out_size;

  {
    int n = M_TOTAL * D_MODEL;
    cvt_bf16_kernel<<<(n + 255) / 256, 256, 0, stream>>>(x, xb, n);
  }
  {
    int n = D_MODEL * N_QKV;
    transpose_cvt_kernel<<<(n + 255) / 256, 256, 0, stream>>>(W_qkv, wqkvT, D_MODEL, N_QKV);
  }
  {
    int n = D_MODEL * D_MODEL;
    transpose_cvt_kernel<<<(n + 255) / 256, 256, 0, stream>>>(W_out, woutT, D_MODEL, D_MODEL);
  }
  {
    // (8192/32) * (3072/64) = 12288 waves, 8 waves/block
    qkv_gemm_kernel<<<12288 / 8, 256, 0, stream>>>(xb, wqkvT, b_qkv, q_ws, k_ws, vt_ws);
  }
  {
    dim3 grid(SEQ / 64, BATCH * N_HEADS);        // 32 x 64, 4 waves/block (16 queries each)
    attn_kernel<<<grid, 128, 0, stream>>>(q_ws, k_ws, vt_ws, y_ws);
  }
  {
    // (8192/32) * (1024/64) = 4096 waves, 8 waves/block
    out_gemm_kernel<<<4096 / 8, 256, 0, stream>>>(y_ws, woutT, b_out, out);
  }
}
